// KOTSinkhornLoss_58394375357197
// MI455X (gfx1250) — matrix operand order repaired and checked by reading, not compiled, not run on hardware
//
#include <hip/hip_runtime.h>
#include <hip/hip_bf16.h>
#include <cstdint>
#include <cstddef>

// ---------------------------------------------------------------------------
// Types for CDNA5 WMMA
// ---------------------------------------------------------------------------
typedef __attribute__((ext_vector_type(16))) _Float16 v16h;
typedef __attribute__((ext_vector_type(8)))  float    v8f;

#define BPTS   2048
#define DIM    1024
#define NLOG2  7.6246189861593985f   /* log(2048) = 11*ln2 */
#define EPSBLR 0.0025f               /* blur^2 = 0.05^2 */
#define NITERS 20

#define BM     128                   /* block tile M */
#define BN     128                   /* block tile N */
#define BK     32                    /* k-step (one WMMA K) */
#define LDPAD  40                    /* halves per LDS row: 32 + 8 pad (80B stride: */
                                     /* 20*row mod 64 banks is a permutation)       */

// ---------------------------------------------------------------------------
// CDNA5 async global->LDS copy (ASYNCcnt-tracked), 16 bytes per lane.
// Syntax per CDNA5 ISA / CDNA5 ISA: VDST = LDS-offset VGPR.
// ---------------------------------------------------------------------------
__device__ inline void async_copy_b128(uint32_t lds_off, const void* gaddr) {
  asm volatile("global_load_async_to_lds_b128 %0, %1, off"
               :: "v"(lds_off), "v"(gaddr)
               : "memory");
}

// ---------------------------------------------------------------------------
// pack: x = [action | sqrt(0.5)*state] -> f16 matrix + row squared norms
// grid: 2048 blocks x 256 threads
// ---------------------------------------------------------------------------
__global__ __launch_bounds__(256) void pack_kernel(
    const float* __restrict__ act, const float* __restrict__ st,
    _Float16* __restrict__ Xh, float* __restrict__ n2) {
  __shared__ float red[256];
  const int row = blockIdx.x;
  const int tid = threadIdx.x;
  float s = 0.0f;
#pragma unroll
  for (int q = 0; q < 4; ++q) {
    const int c = tid * 4 + q;
    float v;
    if (c < 512) v = act[(size_t)row * 512 + c];
    else         v = 0.70710678118654752f * st[(size_t)row * 512 + (c - 512)];
    Xh[(size_t)row * DIM + c] = (_Float16)v;
    s += v * v;
  }
  red[tid] = s;
  __syncthreads();
  for (int o = 128; o > 0; o >>= 1) {
    if (tid < o) red[tid] += red[tid + o];
    __syncthreads();
  }
  if (tid == 0) n2[row] = red[0];
}

// ---------------------------------------------------------------------------
// per-dimension min/max over all 4096 concatenated points (partials)
// grid: dim3(4,16) x 256 threads ; partials[16][1024]
// ---------------------------------------------------------------------------
__global__ __launch_bounds__(256) void minmax_kernel(
    const float* __restrict__ pa, const float* __restrict__ sp,
    const float* __restrict__ ta, const float* __restrict__ stt,
    float* __restrict__ pmin, float* __restrict__ pmax) {
  const int d  = blockIdx.x * 256 + threadIdx.x;   // 0..1023
  const int p0 = blockIdx.y * 256;
  float mn = INFINITY, mx = -INFINITY;
  for (int i = 0; i < 256; ++i) {
    const int p = p0 + i;
    const float* A = (p < BPTS) ? pa : ta;
    const float* S = (p < BPTS) ? sp : stt;
    const int pr = (p < BPTS) ? p : p - BPTS;
    float v;
    if (d < 512) v = A[(size_t)pr * 512 + d];
    else         v = 0.70710678118654752f * S[(size_t)pr * 512 + (d - 512)];
    mn = fminf(mn, v);
    mx = fmaxf(mx, v);
  }
  pmin[blockIdx.y * DIM + d] = mn;
  pmax[blockIdx.y * DIM + d] = mx;
}

// eps0 = diameter^2 = sum_d (max_d - min_d)^2 ; 1 block x 256 threads
__global__ __launch_bounds__(256) void eps0_kernel(
    const float* __restrict__ pmin, const float* __restrict__ pmax,
    float* __restrict__ eps0) {
  __shared__ float red[256];
  const int tid = threadIdx.x;
  float acc = 0.0f;
#pragma unroll
  for (int q = 0; q < 4; ++q) {
    const int d = tid + q * 256;
    float mn = INFINITY, mx = -INFINITY;
    for (int b = 0; b < 16; ++b) {
      mn = fminf(mn, pmin[b * DIM + d]);
      mx = fmaxf(mx, pmax[b * DIM + d]);
    }
    const float df = mx - mn;
    acc += df * df;
  }
  red[tid] = acc;
  __syncthreads();
  for (int o = 128; o > 0; o >>= 1) {
    if (tid < o) red[tid] += red[tid + o];
    __syncthreads();
  }
  if (tid == 0) eps0[0] = red[0];
}

// ---------------------------------------------------------------------------
// Cost GEMM: C[i][j] = 0.5 * max(|x_i|^2 + |y_j|^2 - 2 x_i.y_j, 0)
//
// Block tile 128x128, staged through LDS with double-buffered
// global_load_async_to_lds_b128 (overlapped with WMMA via s_wait_asynccnt).
// 8 waves arranged 4(M) x 2(N): wave tile 32x64 = 2x4 WMMA 16x16 tiles.
// Fragment layouts per CDNA5 ISA §7.12.2:
//   A (16x32 f16): lane<16 -> M=lane, K {0..7,16..23}; lane>=16 -> K {8..15,24..31}
//   B (32x16 f16): N=lane%16, K {0..7,16..23} (lanes<16) / {8..15,24..31} (lanes>=16)
// grid: dim3(2048/128, 2048/128) = (16,16), 256 threads.
// ---------------------------------------------------------------------------
__global__ __launch_bounds__(256) void gemm_cost_kernel(
    const _Float16* __restrict__ Ah, const _Float16* __restrict__ Bh,
    const float* __restrict__ an2, const float* __restrict__ bn2,
    float* __restrict__ Cout, float* __restrict__ CoutT) {
  __shared__ __align__(16) _Float16 sA[2][BM * LDPAD];  // 2 x 10240 B
  __shared__ __align__(16) _Float16 sB[2][BN * LDPAD];  // 2 x 10240 B

  const int tid  = threadIdx.x;
  const int lane = tid & 31;
  const int wave = tid >> 5;        // 0..7
  const int wm   = wave & 3;        // 0..3  -> 32-row slice
  const int wn   = wave >> 2;       // 0..1  -> 64-col slice
  const int rBlk = blockIdx.y * BM;
  const int cBlk = blockIdx.x * BN;

  // Copy assignment: tile is 128 rows x 32 halves = 512 chunks of 8 halves.
  // Each thread stages 2 A-chunks and 2 B-chunks per k-step.
  const int c0   = tid * 2;
  const int row0 = c0 >> 2, k0off = (c0 & 3) * 8;
  const int c1   = c0 + 1;
  const int row1 = c1 >> 2, k1off = (c1 & 3) * 8;

  auto stage = [&](int p, int kbase) {
    async_copy_b128((uint32_t)(uintptr_t)&sA[p][row0 * LDPAD + k0off],
                    Ah + (size_t)(rBlk + row0) * DIM + kbase + k0off);
    async_copy_b128((uint32_t)(uintptr_t)&sA[p][row1 * LDPAD + k1off],
                    Ah + (size_t)(rBlk + row1) * DIM + kbase + k1off);
    async_copy_b128((uint32_t)(uintptr_t)&sB[p][row0 * LDPAD + k0off],
                    Bh + (size_t)(cBlk + row0) * DIM + kbase + k0off);
    async_copy_b128((uint32_t)(uintptr_t)&sB[p][row1 * LDPAD + k1off],
                    Bh + (size_t)(cBlk + row1) * DIM + kbase + k1off);
  };

  v8f acc[2][4];
#pragma unroll
  for (int i = 0; i < 2; ++i)
#pragma unroll
    for (int j = 0; j < 4; ++j) acc[i][j] = (v8f){};

  stage(0, 0);

  const int mrow  = lane & 15;
  const int kselA = (lane < 16) ? 0 : 8;    // halves
  const int kselB = (lane < 16) ? 0 : 8;    // halves

  for (int kk = 0; kk < DIM / BK; ++kk) {
    const int p = kk & 1;
    if (kk + 1 < DIM / BK) {
      stage(p ^ 1, (kk + 1) * BK);
      // wait for this wave's stage-kk copies (leave next stage in flight)
      asm volatile("s_wait_asynccnt 4" ::: "memory");
    } else {
      asm volatile("s_wait_asynccnt 0" ::: "memory");
    }
    __syncthreads();   // all waves' stage-kk copies visible in LDS

    v16h a[2], b[4];
#pragma unroll
    for (int i = 0; i < 2; ++i) {
      const _Float16* pa = &sA[p][(wm * 32 + i * 16 + mrow) * LDPAD + kselA];
      union { v16h v; float4 f[2]; } u;
      u.f[0] = *(const float4*)(pa);
      u.f[1] = *(const float4*)(pa + 16);
      a[i] = u.v;
    }
#pragma unroll
    for (int j = 0; j < 4; ++j) {
      const _Float16* pb = &sB[p][(wn * 64 + j * 16 + mrow) * LDPAD + kselB];
      union { v16h v; float4 f[2]; } u;
      u.f[0] = *(const float4*)(pb);
      u.f[1] = *(const float4*)(pb + 16);
      b[j] = u.v;
    }
#pragma unroll
    for (int i = 0; i < 2; ++i)
#pragma unroll
      for (int j = 0; j < 4; ++j)
        acc[i][j] = __builtin_amdgcn_wmma_f32_16x16x32_f16(
            false, a[i], false, b[j], (short)0, acc[i][j], false, false);

    __syncthreads();   // everyone done reading buffer p before it is re-staged
  }

  // C/D layout: VGPR r -> M = r + (lane<16 ? 0 : 8), N = lane%16
  const int mOff = (lane < 16) ? 0 : 8;
  const int nOff = lane & 15;
#pragma unroll
  for (int i = 0; i < 2; ++i)
#pragma unroll
    for (int j = 0; j < 4; ++j)
#pragma unroll
      for (int r = 0; r < 8; ++r) {
        const int row = rBlk + wm * 32 + i * 16 + mOff + r;
        const int col = cBlk + wn * 64 + j * 16 + nOff;
        const float val =
            0.5f * fmaxf(an2[row] + bn2[col] - 2.0f * acc[i][j][r], 0.0f);
        Cout[(size_t)row * BPTS + col] = val;
        if (CoutT) CoutT[(size_t)col * BPTS + row] = val;
      }
}

// ---------------------------------------------------------------------------
// softmin: out[i] = -eps * logsumexp_j( h_j - C[i][j]/eps ),
//   h_j = -log(B) + (gvec ? gvec[j]/eps : 0)
//   avg: out[i] = 0.5*(f_old[i] + softmin_i)
// eps: t<0 -> eps0 ; 0<=t<20 -> max(eps0*0.25^t, blur^2) ; t>=20 -> blur^2
// One wave per row; block = 8 waves; grid = 256 blocks.
// ---------------------------------------------------------------------------
__global__ __launch_bounds__(256) void softmin_kernel(
    const float* __restrict__ C, const float* __restrict__ gvec,
    const float* __restrict__ f_old, float* __restrict__ out,
    const float* __restrict__ eps0p, int t, int avg) {
  __shared__ float h[BPTS];
  const int tid  = threadIdx.x;
  const int lane = tid & 31;
  const int wave = tid >> 5;

  float eps;
  if (t < 0)            eps = eps0p[0];
  else if (t >= NITERS) eps = EPSBLR;
  else                  eps = fmaxf(eps0p[0] * exp2f(-2.0f * (float)t), EPSBLR);
  const float inv_eps = 1.0f / eps;

  for (int j = tid; j < BPTS; j += 256)
    h[j] = -NLOG2 + (gvec ? gvec[j] * inv_eps : 0.0f);
  __syncthreads();

  const int row = blockIdx.x * 8 + wave;
  const float4* crow = (const float4*)(C + (size_t)row * BPTS);

  float4 vals[16];
  float m = -INFINITY;
#pragma unroll
  for (int c = 0; c < 16; ++c) {
    const float4 cv = crow[c * 32 + lane];
    const float4 hv = *(const float4*)&h[c * 128 + lane * 4];
    float4 v;
    v.x = hv.x - cv.x * inv_eps;
    v.y = hv.y - cv.y * inv_eps;
    v.z = hv.z - cv.z * inv_eps;
    v.w = hv.w - cv.w * inv_eps;
    vals[c] = v;
    m = fmaxf(m, fmaxf(fmaxf(v.x, v.y), fmaxf(v.z, v.w)));
  }
#pragma unroll
  for (int o = 16; o > 0; o >>= 1) m = fmaxf(m, __shfl_xor(m, o, 32));

  float s = 0.0f;
#pragma unroll
  for (int c = 0; c < 16; ++c) {
    const float4 v = vals[c];
    s += __expf(v.x - m) + __expf(v.y - m) + __expf(v.z - m) + __expf(v.w - m);
  }
#pragma unroll
  for (int o = 16; o > 0; o >>= 1) s += __shfl_xor(s, o, 32);

  if (lane == 0) {
    float r = -eps * (m + __logf(s));
    if (avg) r = 0.5f * (f_old[row] + r);
    out[row] = r;
  }
}

// loss = (1/B) * sum_i (fba_n - faa_n) + (1/B) * sum_j (gab_n - gbb_n)
__global__ __launch_bounds__(256) void loss_kernel(
    const float* __restrict__ fban, const float* __restrict__ faan,
    const float* __restrict__ gabn, const float* __restrict__ gbbn,
    float* __restrict__ out) {
  __shared__ float red[256];
  const int tid = threadIdx.x;
  float s = 0.0f;
  for (int j = tid; j < BPTS; j += 256)
    s += (fban[j] - faan[j]) + (gabn[j] - gbbn[j]);
  red[tid] = s;
  __syncthreads();
  for (int o = 128; o > 0; o >>= 1) {
    if (tid < o) red[tid] += red[tid + o];
    __syncthreads();
  }
  if (tid == 0) out[0] = red[0] * (1.0f / (float)BPTS);
}

// ---------------------------------------------------------------------------
extern "C" void kernel_launch(void* const* d_in, const int* in_sizes, int n_in,
                              void* d_out, int out_size, void* d_ws,
                              size_t ws_size, hipStream_t stream) {
  (void)in_sizes; (void)n_in; (void)out_size; (void)ws_size;
  const float* pa  = (const float*)d_in[0];  // pred_action  (2048,16,32)
  const float* ta  = (const float*)d_in[1];  // target_action
  const float* sp  = (const float*)d_in[2];  // state_pred   (2048,512)
  const float* stt = (const float*)d_in[3];  // state_target

  uint8_t* ws = (uint8_t*)d_ws;
  size_t off = 0;
  auto nxt = [&](size_t bytes) -> void* {
    void* p = (void*)(ws + off);
    off += (bytes + 255) & ~(size_t)255;
    return p;
  };

  _Float16* xh  = (_Float16*)nxt((size_t)BPTS * DIM * 2);   // 4 MB
  _Float16* yh  = (_Float16*)nxt((size_t)BPTS * DIM * 2);   // 4 MB
  float* Cxy = (float*)nxt((size_t)BPTS * BPTS * 4);        // 16 MB
  float* Cyx = (float*)nxt((size_t)BPTS * BPTS * 4);        // 16 MB
  float* Cxx = (float*)nxt((size_t)BPTS * BPTS * 4);        // 16 MB
  float* Cyy = (float*)nxt((size_t)BPTS * BPTS * 4);        // 16 MB
  float* x2   = (float*)nxt(BPTS * 4);
  float* y2   = (float*)nxt(BPTS * 4);
  float* pmin = (float*)nxt(16 * DIM * 4);
  float* pmax = (float*)nxt(16 * DIM * 4);
  float* eps0 = (float*)nxt(256);
  float* fba[2], *gab[2], *faa[2], *gbb[2];
  for (int i = 0; i < 2; ++i) {
    fba[i] = (float*)nxt(BPTS * 4);
    gab[i] = (float*)nxt(BPTS * 4);
    faa[i] = (float*)nxt(BPTS * 4);
    gbb[i] = (float*)nxt(BPTS * 4);
  }
  float* fban = (float*)nxt(BPTS * 4);
  float* gabn = (float*)nxt(BPTS * 4);
  float* faan = (float*)nxt(BPTS * 4);
  float* gbbn = (float*)nxt(BPTS * 4);
  // total scratch ~ 76 MB

  // 1) assemble f16 point matrices + squared norms
  pack_kernel<<<BPTS, 256, 0, stream>>>(pa,  sp,  xh, x2);
  pack_kernel<<<BPTS, 256, 0, stream>>>(ta,  stt, yh, y2);

  // 2) diameter^2 -> eps0
  minmax_kernel<<<dim3(4, 16), 256, 0, stream>>>(pa, sp, ta, stt, pmin, pmax);
  eps0_kernel<<<1, 256, 0, stream>>>(pmin, pmax, eps0);

  // 3) cost matrices via WMMA GEMM (C_yx stored explicitly as transpose)
  const dim3 gg(BPTS / BN, BPTS / BM);
  gemm_cost_kernel<<<gg, 256, 0, stream>>>(xh, yh, x2, y2, Cxy, Cyx);
  gemm_cost_kernel<<<gg, 256, 0, stream>>>(xh, xh, x2, x2, Cxx, nullptr);
  gemm_cost_kernel<<<gg, 256, 0, stream>>>(yh, yh, y2, y2, Cyy, nullptr);

  // 4) init potentials at eps0
  softmin_kernel<<<256, 256, 0, stream>>>(Cxx, nullptr, nullptr, faa[0], eps0, -1, 0);
  softmin_kernel<<<256, 256, 0, stream>>>(Cyy, nullptr, nullptr, gbb[0], eps0, -1, 0);
  softmin_kernel<<<256, 256, 0, stream>>>(Cyx, nullptr, nullptr, gab[0], eps0, -1, 0);
  softmin_kernel<<<256, 256, 0, stream>>>(Cxy, nullptr, nullptr, fba[0], eps0, -1, 0);

  // 5) annealed symmetric Sinkhorn loop (all reads from buffer r, writes to w)
  for (int t = 0; t < NITERS; ++t) {
    const int r = t & 1, w = 1 - r;
    softmin_kernel<<<256, 256, 0, stream>>>(Cxy, gab[r], fba[r], fba[w], eps0, t, 1);
    softmin_kernel<<<256, 256, 0, stream>>>(Cyx, fba[r], gab[r], gab[w], eps0, t, 1);
    softmin_kernel<<<256, 256, 0, stream>>>(Cxx, faa[r], faa[r], faa[w], eps0, t, 1);
    softmin_kernel<<<256, 256, 0, stream>>>(Cyy, gbb[r], gbb[r], gbb[w], eps0, t, 1);
  }

  // 6) final extrapolation at eps = blur^2 (carry sits in buffer index 0)
  const int f = NITERS & 1 ? 1 : 0;
  softmin_kernel<<<256, 256, 0, stream>>>(Cxy, gab[f], nullptr, fban, eps0, NITERS, 0);
  softmin_kernel<<<256, 256, 0, stream>>>(Cyx, fba[f], nullptr, gabn, eps0, NITERS, 0);
  softmin_kernel<<<256, 256, 0, stream>>>(Cxx, faa[f], nullptr, faan, eps0, NITERS, 0);
  softmin_kernel<<<256, 256, 0, stream>>>(Cyy, gbb[f], nullptr, gbbn, eps0, NITERS, 0);

  // 7) assemble scalar loss
  loss_kernel<<<1, 256, 0, stream>>>(fban, faan, gabn, gbbn, (float*)d_out);
}
